// bPC_SNN_59631325937986
// MI455X (gfx1250) — matrix operand order repaired and checked by reading, not compiled, not run on hardware
//
#include <hip/hip_runtime.h>

typedef __bf16 bf16_t;
typedef __attribute__((ext_vector_type(16))) __bf16 v16bf;
typedef __attribute__((ext_vector_type(8)))  __bf16 v8bf;
typedef __attribute__((ext_vector_type(8)))  float  v8f;

typedef unsigned long long u64;

// ---------------- workspace layout (byte offsets, 256-aligned) ----------------
constexpr u64 AL(u64 x) { return (x + 255ull) & ~255ull; }

constexpr u64 E_W0  = 1024ull * 2048;   // W0/V0-sized weight (elems)
constexpr u64 E_WH  = 2048ull * 2048;   // hidden weight
constexpr u64 E_B1K = 256ull * 1024;    // batch x 1024
constexpr u64 E_B2K = 256ull * 2048;    // batch x 2048
constexpr u64 E_BL  = 256ull * 10;      // batch x 10

// bf16 weights (built once per call)
constexpr u64 OFF_W0b  = 0;
constexpr u64 OFF_W0Tb = AL(OFF_W0b  + E_W0 * 2);
constexpr u64 OFF_W1b  = AL(OFF_W0Tb + E_W0 * 2);
constexpr u64 OFF_W1Tb = AL(OFF_W1b  + E_WH * 2);
constexpr u64 OFF_W2b  = AL(OFF_W1Tb + E_WH * 2);
constexpr u64 OFF_W2Tb = AL(OFF_W2b  + E_WH * 2);
constexpr u64 OFF_V1b  = AL(OFF_W2Tb + E_WH * 2);
constexpr u64 OFF_V1Tb = AL(OFF_V1b  + E_WH * 2);
constexpr u64 OFF_V2b  = AL(OFF_V1Tb + E_WH * 2);
constexpr u64 OFF_V2Tb = AL(OFF_V2b  + E_WH * 2);
constexpr u64 OFF_V0b  = AL(OFF_V2Tb + E_WH * 2);
constexpr u64 OFF_XDb  = AL(OFF_V0b  + E_W0 * 2);   // x_data bf16
// fp32 time-invariant matmuls
constexpr u64 OFF_CV0  = AL(OFF_XDb + E_B1K * 2);   // x_data @ V0^T  [256,2048]
constexpr u64 OFF_CW3  = AL(OFF_CV0 + E_B2K * 4);   // y @ W3^T       [256,2048]
// --------- zeroed state region ---------
constexpr u64 OFF_STATE = AL(OFF_CW3 + E_B2K * 4);
constexpr u64 OFF_S0   = OFF_STATE;                 // spikes (bf16)
constexpr u64 OFF_S1A  = AL(OFF_S0  + E_B1K * 2);
constexpr u64 OFF_S1B  = AL(OFF_S1A + E_B2K * 2);
constexpr u64 OFF_S2   = AL(OFF_S1B + E_B2K * 2);
constexpr u64 OFF_S3   = AL(OFF_S2  + E_B2K * 2);
constexpr u64 OFF_EG0  = AL(OFF_S3  + E_B2K * 2);   // errors (bf16)
constexpr u64 OFF_EG1  = AL(OFF_EG0 + E_B1K * 2);
constexpr u64 OFF_EG2  = AL(OFF_EG1 + E_B2K * 2);
constexpr u64 OFF_EG3  = AL(OFF_EG2 + E_B2K * 2);
constexpr u64 OFF_ED1  = AL(OFF_EG3 + E_B2K * 2);
constexpr u64 OFF_ED2  = AL(OFF_ED1 + E_B2K * 2);
constexpr u64 OFF_ED3  = AL(OFF_ED2 + E_B2K * 2);
constexpr u64 OFF_ED4  = AL(OFF_ED3 + E_B2K * 2);
constexpr u64 OFF_S4   = AL(OFF_ED4 + E_BL * 2);
constexpr u64 OFF_MV0  = AL(OFF_S4  + E_BL * 2);    // fp32 membrane/current/trace
constexpr u64 OFF_MJ0  = AL(OFF_MV0 + E_B1K * 4);
constexpr u64 OFF_MV1  = AL(OFF_MJ0 + E_B1K * 4);
constexpr u64 OFF_MJ1  = AL(OFF_MV1 + E_B2K * 4);
constexpr u64 OFF_MX1  = AL(OFF_MJ1 + E_B2K * 4);
constexpr u64 OFF_MV2  = AL(OFF_MX1 + E_B2K * 4);
constexpr u64 OFF_MJ2  = AL(OFF_MV2 + E_B2K * 4);
constexpr u64 OFF_MX2  = AL(OFF_MJ2 + E_B2K * 4);
constexpr u64 OFF_MV3  = AL(OFF_MX2 + E_B2K * 4);
constexpr u64 OFF_MJ3  = AL(OFF_MV3 + E_B2K * 4);
constexpr u64 OFF_MX3  = AL(OFF_MJ3 + E_B2K * 4);
constexpr u64 OFF_MV4  = AL(OFF_MX3 + E_B2K * 4);
constexpr u64 OFF_MJ4  = AL(OFF_MV4 + E_BL * 4);
constexpr u64 OFF_SC   = AL(OFF_MJ4 + E_BL * 4);
constexpr u64 OFF_VS   = AL(OFF_SC  + E_BL * 4);
constexpr u64 OFF_END  = AL(OFF_VS  + E_BL * 4);

union Frag { v16bf v; v8bf h[2]; };

// ---------------- WMMA GEMM core: wave computes 32x64 of C -------------------
// A: [M,K] row-major bf16 (lda). Bm: [N,K] row-major bf16 (ldb). fp32 accum.
__device__ __forceinline__ void gemm_accum(
    const bf16_t* __restrict__ A, int lda,
    const bf16_t* __restrict__ Bm, int ldb, int K,
    int m0, int n0, int lane, v8f (&acc)[2][4]) {
  const int half = lane >> 4, r = lane & 15;
  for (int k0 = 0; k0 < K; k0 += 32) {
    Frag a0, a1;
    {
      const bf16_t* p0 = A + (u64)(m0 + r) * lda + k0 + half * 8;
      a0.h[0] = *(const v8bf*)p0;
      a0.h[1] = *(const v8bf*)(p0 + 16);
      const bf16_t* p1 = A + (u64)(m0 + 16 + r) * lda + k0 + half * 8;
      a1.h[0] = *(const v8bf*)p1;
      a1.h[1] = *(const v8bf*)(p1 + 16);
    }
#pragma unroll
    for (int nt = 0; nt < 4; ++nt) {
      Frag b;
      const bf16_t* pb = Bm + (u64)(n0 + nt * 16 + r) * ldb + k0 + half * 8;
      b.h[0] = *(const v8bf*)pb;
      b.h[1] = *(const v8bf*)(pb + 16);
      acc[0][nt] = __builtin_amdgcn_wmma_f32_16x16x32_bf16(
          false, a0.v, false, b.v, (short)0, acc[0][nt], false, false);
      acc[1][nt] = __builtin_amdgcn_wmma_f32_16x16x32_bf16(
          false, a1.v, false, b.v, (short)0, acc[1][nt], false, false);
    }
  }
}

__device__ __forceinline__ void zero_acc(v8f (&acc)[2][4]) {
  const v8f z = {0.f, 0.f, 0.f, 0.f, 0.f, 0.f, 0.f, 0.f};
#pragma unroll
  for (int i = 0; i < 2; ++i)
#pragma unroll
    for (int j = 0; j < 4; ++j) acc[i][j] = z;
}

// fused LIF epilogue over the wave's 32x64 C tile
__device__ __forceinline__ void epi_lif(
    v8f (&acc)[2][4], int m0, int n0, int lane, int N,
    const bf16_t* ed, const bf16_t* eg,          // -ed - eg input terms (or null)
    const bf16_t* ed4, const float* V3f,         // + ed4 @ V3 (K=10) term (or null)
    float* MV, float* MJ, float* MX, bf16_t* S) {
  const int half = lane >> 4, r = lane & 15;
#pragma unroll
  for (int mt = 0; mt < 2; ++mt)
#pragma unroll
    for (int nt = 0; nt < 4; ++nt)
#pragma unroll
      for (int p = 0; p < 8; ++p) {
        const int m = m0 + mt * 16 + p + half * 8;
        const int n = n0 + nt * 16 + r;
        const u64 o = (u64)m * N + n;
        float inp = acc[mt][nt][p];
        if (ed4) {
#pragma unroll
          for (int k = 0; k < 10; ++k)
            inp += (float)ed4[m * 10 + k] * V3f[k * 2048 + n];
        }
        if (ed) inp -= (float)ed[o];
        if (eg) inp -= (float)eg[o];
        float j = 0.9f * MJ[o] + 0.1f * inp;   // dt/tau_j = 0.1
        float v = 0.95f * MV[o] + 0.05f * j;   // dt/tau_m = 0.05
        float s = (v > 1.0f) ? 1.0f : 0.0f;
        v *= (1.0f - s);
        MV[o] = v; MJ[o] = j; S[o] = (bf16_t)s;
        if (MX) MX[o] = 0.95f * MX[o] + s;     // 1 - 1/tau_tr
      }
}

// ---------------- kernels ----------------------------------------------------
__global__ void k_zero(unsigned* __restrict__ p, u64 nwords) {
  for (u64 i = (u64)blockIdx.x * blockDim.x + threadIdx.x; i < nwords;
       i += (u64)gridDim.x * blockDim.x)
    p[i] = 0u;
}

__global__ void k_cvt(const float* __restrict__ s, bf16_t* __restrict__ d, int n) {
  for (int i = blockIdx.x * blockDim.x + threadIdx.x; i < n;
       i += gridDim.x * blockDim.x)
    d[i] = (bf16_t)s[i];
}

// src fp32 [R][C]  ->  dst bf16 [C][R]
__global__ void k_tr(const float* __restrict__ src, bf16_t* __restrict__ dst,
                     int R, int C) {
  __shared__ float tile[32][33];
  const int cb = blockIdx.x * 32, rb = blockIdx.y * 32;
  const int tx = threadIdx.x, ty = threadIdx.y;  // (32,8)
  for (int i = ty; i < 32; i += 8)
    tile[i][tx] = src[(u64)(rb + i) * C + cb + tx];
  __syncthreads();
  for (int i = ty; i < 32; i += 8)
    dst[(u64)(cb + i) * R + rb + tx] = (bf16_t)tile[tx][i];
}

// CW3[b][n] = sum_{k<10} y[b][k] * W3[n][k]     (time-invariant)
__global__ void k_cw3(const float* __restrict__ y, const float* __restrict__ W3,
                      float* __restrict__ cw3) {
  const int idx = blockIdx.x * 256 + threadIdx.x;   // 256*2048
  const int b = idx >> 11, n = idx & 2047;
  float a = 0.f;
#pragma unroll
  for (int k = 0; k < 10; ++k) a += y[b * 10 + k] * W3[n * 10 + k];
  cw3[idx] = a;
}

// CV0 = x_data @ V0^T  (time-invariant; WMMA, fp32 store)
__global__ void __launch_bounds__(256) k_cv0(char* __restrict__ ws) {
  const int lane = threadIdx.x & 31, wave = threadIdx.x >> 5;
  const int wm = wave & 3, wn = wave >> 2;
  const int mb = blockIdx.x >> 4, nb = blockIdx.x & 15;
  const int m0 = mb * 128 + wm * 32, n0 = nb * 128 + wn * 64;
  v8f acc[2][4]; zero_acc(acc);
  gemm_accum((const bf16_t*)(ws + OFF_XDb), 1024,
             (const bf16_t*)(ws + OFF_V0b), 1024, 1024, m0, n0, lane, acc);
  float* C = (float*)(ws + OFF_CV0);
  const int half = lane >> 4, r = lane & 15;
#pragma unroll
  for (int mt = 0; mt < 2; ++mt)
#pragma unroll
    for (int nt = 0; nt < 4; ++nt)
#pragma unroll
      for (int p = 0; p < 8; ++p)
        C[(u64)(m0 + mt * 16 + p + half * 8) * 2048 + n0 + nt * 16 + r] =
            acc[mt][nt][p];
}

// Phase A: layers 0..3 update (all read OLD state; s1 double-buffered)
__global__ void __launch_bounds__(256) k_phaseA(char* __restrict__ ws, int parity,
                                                const float* __restrict__ V3f) {
  const int lane = threadIdx.x & 31, wave = threadIdx.x >> 5;
  const int wm = wave & 3, wn = wave >> 2;
  const int bid = blockIdx.x;
  v8f acc[2][4]; zero_acc(acc);

  bf16_t* S1w = (bf16_t*)(ws + (parity ? OFF_S1B : OFF_S1A));
  const bf16_t* S1o = (const bf16_t*)(ws + (parity ? OFF_S1A : OFF_S1B));

  if (bid < 16) {                       // layer 0: inp = s1_old @ W0^T, N=1024
    const int mb = bid >> 3, nb = bid & 7;
    const int m0 = mb * 128 + wm * 32, n0 = nb * 128 + wn * 64;
    gemm_accum(S1o, 2048, (const bf16_t*)(ws + OFF_W0b), 2048, 2048, m0, n0, lane, acc);
    epi_lif(acc, m0, n0, lane, 1024, nullptr, nullptr, nullptr, nullptr,
            (float*)(ws + OFF_MV0), (float*)(ws + OFF_MJ0), nullptr,
            (bf16_t*)(ws + OFF_S0));
  } else if (bid < 48) {                // layer 1: eg0@W0 + ed2@V1 - ed1 - eg1
    const int q = bid - 16, mb = q >> 4, nb = q & 15;
    const int m0 = mb * 128 + wm * 32, n0 = nb * 128 + wn * 64;
    gemm_accum((const bf16_t*)(ws + OFF_EG0), 1024,
               (const bf16_t*)(ws + OFF_W0Tb), 1024, 1024, m0, n0, lane, acc);
    gemm_accum((const bf16_t*)(ws + OFF_ED2), 2048,
               (const bf16_t*)(ws + OFF_V1Tb), 2048, 2048, m0, n0, lane, acc);
    epi_lif(acc, m0, n0, lane, 2048,
            (const bf16_t*)(ws + OFF_ED1), (const bf16_t*)(ws + OFF_EG1),
            nullptr, nullptr,
            (float*)(ws + OFF_MV1), (float*)(ws + OFF_MJ1), (float*)(ws + OFF_MX1),
            S1w);
  } else if (bid < 80) {                // layer 2: eg1@W1 + ed3@V2 - ed2 - eg2
    const int q = bid - 48, mb = q >> 4, nb = q & 15;
    const int m0 = mb * 128 + wm * 32, n0 = nb * 128 + wn * 64;
    gemm_accum((const bf16_t*)(ws + OFF_EG1), 2048,
               (const bf16_t*)(ws + OFF_W1Tb), 2048, 2048, m0, n0, lane, acc);
    gemm_accum((const bf16_t*)(ws + OFF_ED3), 2048,
               (const bf16_t*)(ws + OFF_V2Tb), 2048, 2048, m0, n0, lane, acc);
    epi_lif(acc, m0, n0, lane, 2048,
            (const bf16_t*)(ws + OFF_ED2), (const bf16_t*)(ws + OFF_EG2),
            nullptr, nullptr,
            (float*)(ws + OFF_MV2), (float*)(ws + OFF_MJ2), (float*)(ws + OFF_MX2),
            (bf16_t*)(ws + OFF_S2));
  } else {                              // layer 3: eg2@W2 + ed4@V3 - ed3 - eg3
    const int q = bid - 80, mb = q >> 4, nb = q & 15;
    const int m0 = mb * 128 + wm * 32, n0 = nb * 128 + wn * 64;
    gemm_accum((const bf16_t*)(ws + OFF_EG2), 2048,
               (const bf16_t*)(ws + OFF_W2Tb), 2048, 2048, m0, n0, lane, acc);
    epi_lif(acc, m0, n0, lane, 2048,
            (const bf16_t*)(ws + OFF_ED3), (const bf16_t*)(ws + OFF_EG3),
            (const bf16_t*)(ws + OFF_ED4), V3f,
            (float*)(ws + OFF_MV3), (float*)(ws + OFF_MJ3), (float*)(ws + OFF_MX3),
            (bf16_t*)(ws + OFF_S3));
  }
}

// Phase B: label layer (fresh s3), accumulates readout, sets ed4
__global__ void k_label(char* __restrict__ ws, const float* __restrict__ y,
                        const float* __restrict__ V3f) {
  const int idx = blockIdx.x * 256 + threadIdx.x;   // 2560
  if (idx >= 2560) return;
  const int b = idx / 10, n = idx % 10;
  const bf16_t* s3 = (const bf16_t*)(ws + OFF_S3);
  float a = 0.f;
  for (int k = 0; k < 2048; ++k)
    a += (float)s3[b * 2048 + k] * V3f[n * 2048 + k];
  float* MV4 = (float*)(ws + OFF_MV4);
  float* MJ4 = (float*)(ws + OFF_MJ4);
  float j = 0.9f * MJ4[idx] + 0.1f * a;
  float v = 0.95f * MV4[idx] + 0.05f * j;
  float s = (v > 1.0f) ? 1.0f : 0.0f;
  v *= (1.0f - s);
  MV4[idx] = v; MJ4[idx] = j;
  ((bf16_t*)(ws + OFF_S4))[idx]  = (bf16_t)s;
  ((bf16_t*)(ws + OFF_ED4))[idx] = (bf16_t)(0.5f * (y[idx] - s));
  ((float*)(ws + OFF_SC))[idx] += s;
  ((float*)(ws + OFF_VS))[idx] += v;
}

// Phase B: four error GEMMs (fresh s/x): eg1, ed2, eg2, ed3
__global__ void __launch_bounds__(256) k_err(char* __restrict__ ws, int parity) {
  const int lane = threadIdx.x & 31, wave = threadIdx.x >> 5;
  const int wm = wave & 3, wn = wave >> 2;
  const int g = blockIdx.x >> 5, q = blockIdx.x & 31;
  const int mb = q >> 4, nb = q & 15;
  const int m0 = mb * 128 + wm * 32, n0 = nb * 128 + wn * 64;
  const bf16_t* S1c = (const bf16_t*)(ws + (parity ? OFF_S1B : OFF_S1A));
  const bf16_t* A; const bf16_t* Bm; const float* X; bf16_t* D;
  switch (g) {
    case 0:  A = (const bf16_t*)(ws + OFF_S2); Bm = (const bf16_t*)(ws + OFF_W1b);
             X = (const float*)(ws + OFF_MX1); D = (bf16_t*)(ws + OFF_EG1); break;
    case 1:  A = S1c;                          Bm = (const bf16_t*)(ws + OFF_V1b);
             X = (const float*)(ws + OFF_MX2); D = (bf16_t*)(ws + OFF_ED2); break;
    case 2:  A = (const bf16_t*)(ws + OFF_S3); Bm = (const bf16_t*)(ws + OFF_W2b);
             X = (const float*)(ws + OFF_MX2); D = (bf16_t*)(ws + OFF_EG2); break;
    default: A = (const bf16_t*)(ws + OFF_S2); Bm = (const bf16_t*)(ws + OFF_V2b);
             X = (const float*)(ws + OFF_MX3); D = (bf16_t*)(ws + OFF_ED3); break;
  }
  v8f acc[2][4]; zero_acc(acc);
  gemm_accum(A, 2048, Bm, 2048, 2048, m0, n0, lane, acc);
  const int half = lane >> 4, r = lane & 15;
#pragma unroll
  for (int mt = 0; mt < 2; ++mt)
#pragma unroll
    for (int nt = 0; nt < 4; ++nt)
#pragma unroll
      for (int p = 0; p < 8; ++p) {
        const u64 o = (u64)(m0 + mt * 16 + p + half * 8) * 2048 + n0 + nt * 16 + r;
        D[o] = (bf16_t)(0.5f * (X[o] - acc[mt][nt][p]));
      }
}

// Phase B: elementwise errors eg0, ed1, eg3
__global__ void k_elem(char* __restrict__ ws, const float* __restrict__ xd) {
  const int i = blockIdx.x * 256 + threadIdx.x;   // 1,310,720 total
  if (i < 262144) {
    const bf16_t* S0 = (const bf16_t*)(ws + OFF_S0);
    ((bf16_t*)(ws + OFF_EG0))[i] = (bf16_t)(0.5f * (xd[i] - (float)S0[i]));
  } else if (i < 786432) {
    const int j = i - 262144;
    ((bf16_t*)(ws + OFF_ED1))[j] =
        (bf16_t)(0.5f * (((const float*)(ws + OFF_MX1))[j] -
                         ((const float*)(ws + OFF_CV0))[j]));
  } else {
    const int j = i - 786432;
    ((bf16_t*)(ws + OFF_EG3))[j] =
        (bf16_t)(0.5f * (((const float*)(ws + OFF_MX3))[j] -
                         ((const float*)(ws + OFF_CW3))[j]));
  }
}

__global__ void k_out(const char* __restrict__ ws, float* __restrict__ out) {
  const int i = blockIdx.x * 256 + threadIdx.x;
  if (i < 2560) {
    out[i]        = ((const float*)(ws + OFF_SC))[i];
    out[2560 + i] = ((const float*)(ws + OFF_VS))[i];
  }
}

// ---------------- host launch -------------------------------------------------
extern "C" void kernel_launch(void* const* d_in, const int* in_sizes, int n_in,
                              void* d_out, int out_size, void* d_ws, size_t ws_size,
                              hipStream_t stream) {
  (void)in_sizes; (void)n_in; (void)out_size;
  if (ws_size < OFF_END) return;   // workspace too small: leave output untouched
  char* ws = (char*)d_ws;
  const float* xd = (const float*)d_in[0];
  const float* yt = (const float*)d_in[1];
  const float* W0 = (const float*)d_in[2];
  const float* V0 = (const float*)d_in[3];
  const float* W1 = (const float*)d_in[4];
  const float* V1 = (const float*)d_in[5];
  const float* W2 = (const float*)d_in[6];
  const float* V2 = (const float*)d_in[7];
  const float* W3 = (const float*)d_in[8];
  const float* V3 = (const float*)d_in[9];
  float* out = (float*)d_out;

  // zero state region
  k_zero<<<1024, 256, 0, stream>>>((unsigned*)(ws + OFF_STATE),
                                   (OFF_END - OFF_STATE) / 4);
  // bf16 weight copies
  k_cvt<<<1024, 256, 0, stream>>>(W0, (bf16_t*)(ws + OFF_W0b), (int)E_W0);
  k_cvt<<<1024, 256, 0, stream>>>(V0, (bf16_t*)(ws + OFF_V0b), (int)E_W0);
  k_cvt<<<1024, 256, 0, stream>>>(W1, (bf16_t*)(ws + OFF_W1b), (int)E_WH);
  k_cvt<<<1024, 256, 0, stream>>>(V1, (bf16_t*)(ws + OFF_V1b), (int)E_WH);
  k_cvt<<<1024, 256, 0, stream>>>(W2, (bf16_t*)(ws + OFF_W2b), (int)E_WH);
  k_cvt<<<1024, 256, 0, stream>>>(V2, (bf16_t*)(ws + OFF_V2b), (int)E_WH);
  k_cvt<<<1024, 256, 0, stream>>>(xd, (bf16_t*)(ws + OFF_XDb), (int)E_B1K);
  // bf16 transposed copies
  dim3 tb(32, 8);
  k_tr<<<dim3(2048 / 32, 1024 / 32), tb, 0, stream>>>(W0, (bf16_t*)(ws + OFF_W0Tb), 1024, 2048);
  k_tr<<<dim3(2048 / 32, 2048 / 32), tb, 0, stream>>>(W1, (bf16_t*)(ws + OFF_W1Tb), 2048, 2048);
  k_tr<<<dim3(2048 / 32, 2048 / 32), tb, 0, stream>>>(W2, (bf16_t*)(ws + OFF_W2Tb), 2048, 2048);
  k_tr<<<dim3(2048 / 32, 2048 / 32), tb, 0, stream>>>(V1, (bf16_t*)(ws + OFF_V1Tb), 2048, 2048);
  k_tr<<<dim3(2048 / 32, 2048 / 32), tb, 0, stream>>>(V2, (bf16_t*)(ws + OFF_V2Tb), 2048, 2048);
  // time-invariant products
  k_cv0<<<32, 256, 0, stream>>>(ws);
  k_cw3<<<2048, 256, 0, stream>>>(yt, W3, (float*)(ws + OFF_CW3));

  // 64 timesteps: phase A (layers 0-3), then phase B (label + errors)
  for (int t = 0; t < 64; ++t) {
    const int p = t & 1;
    k_phaseA<<<112, 256, 0, stream>>>(ws, p, V3);
    k_label<<<10, 256, 0, stream>>>(ws, yt, V3);
    k_err<<<128, 256, 0, stream>>>(ws, p);
    k_elem<<<5120, 256, 0, stream>>>(ws, xd);
  }
  k_out<<<10, 256, 0, stream>>>(ws, out);
}
